// RWKV_6176162972110
// MI455X (gfx1250) — matrix operand order repaired and checked by reading, not compiled, not run on hardware
//
#include <hip/hip_runtime.h>
#include <math.h>

// ---------------- problem constants (RWKV: L=24, D=1024, F=4096, V=50277) ---
#define D_EMB   1024
#define F_HID   4096
#define N_LAYER 24
#define VOCAB   50277

typedef __attribute__((ext_vector_type(16))) _Float16 v16h;
typedef __attribute__((ext_vector_type(8)))  float    v8f;
typedef __attribute__((ext_vector_type(2)))  __fp16   h2;   // matches cvt_pkrtz return

// ---------------------------------------------------------------------------
// 256-thread block sum reduction
// ---------------------------------------------------------------------------
__device__ __forceinline__ float block_sum256(float v, float* sm) {
    const int t = threadIdx.x;
    sm[t] = v;
    __syncthreads();
    for (int s = 128; s > 0; s >>= 1) {
        if (t < s) sm[t] += sm[t + s];
        __syncthreads();
    }
    float r = sm[0];
    __syncthreads();
    return r;
}

// ---------------------------------------------------------------------------
// LayerNorm + (optional) token-shift mixing.
//   xn = LN(x, w, b);  xn -> xn_out (state row) if non-null
//   x{k,v,r}[d] = xn*tm[d] + st[d]*(1-tm[d])   (each optional)
// Launch: 1 block x 256 threads.
// ---------------------------------------------------------------------------
__global__ void k_ln_mix(const float* __restrict__ x, int n,
                         const float* __restrict__ w, const float* __restrict__ b,
                         const float* __restrict__ st,
                         const float* __restrict__ tk, const float* __restrict__ tv,
                         const float* __restrict__ tr,
                         float* __restrict__ xn_out,
                         float* __restrict__ xk, float* __restrict__ xv,
                         float* __restrict__ xr) {
    __shared__ float sm[256];
    const int tid = threadIdx.x;

    float acc = 0.f;
    for (int d = tid; d < n; d += 256) acc += x[d];
    const float mean = block_sum256(acc, sm) / (float)n;

    float vacc = 0.f;
    for (int d = tid; d < n; d += 256) { float t = x[d] - mean; vacc += t * t; }
    const float var  = block_sum256(vacc, sm) / (float)n;
    const float rstd = rsqrtf(var + 1e-5f);

    for (int d = tid; d < n; d += 256) {
        const float xn = (x[d] - mean) * rstd * w[d] + b[d];
        if (xn_out) xn_out[d] = xn;
        const float sa = st ? st[d] : 0.f;
        if (xk) { const float m = tk[d]; xk[d] = xn * m + sa * (1.f - m); }
        if (xv) { const float m = tv[d]; xv[d] = xn * m + sa * (1.f - m); }
        if (xr) { const float m = tr[d]; xr[d] = xn * m + sa * (1.f - m); }
    }
}

// ---------------------------------------------------------------------------
// WKV recurrence (elementwise over D). r is already sigmoided.
// Writes t_out = r * (a/b) and new state rows (aa, bb, p2).
// ---------------------------------------------------------------------------
__global__ void k_wkv(const float* __restrict__ kk, const float* __restrict__ vv,
                      const float* __restrict__ r,
                      const float* __restrict__ tf, const float* __restrict__ td,
                      const float* __restrict__ aa, const float* __restrict__ bb,
                      const float* __restrict__ pp,
                      float* __restrict__ t_out, float* __restrict__ naa,
                      float* __restrict__ nbb, float* __restrict__ np2) {
    for (int d = threadIdx.x; d < D_EMB; d += 256) {
        const float k  = kk[d], v = vv[d], ppd = pp[d], aad = aa[d], bbd = bb[d];
        const float ww = tf[d] + k;
        const float p  = fmaxf(ppd, ww);
        const float e1 = expf(ppd - p), e2 = expf(ww - p);
        const float a  = e1 * aad + e2 * v;
        const float bq = e1 * bbd + e2;
        const float ww2 = ppd + td[d];
        const float p2  = fmaxf(ww2, k);
        const float f1 = expf(ww2 - p2), f2 = expf(k - p2);
        naa[d] = f1 * aad + f2 * v;
        nbb[d] = f1 * bbd + f2;
        np2[d] = p2;
        t_out[d] = r[d] * (a / bq);
    }
}

// ---------------------------------------------------------------------------
// WMMA GEMV:  y[row] = op( dot(W[row,:], x) )   for a 16-row tile per block.
// W streamed as fp16 A-operand (100% memory efficiency), x replicated across
// the 16 B columns, fp32 accumulation. 8 wave32s split K; LDS reduce.
// NCOLS/MODE are compile-time -> K loop unrolls, loads clause, no loop SALU.
// MODE: 0 y=s | 1 y+=s | 2 y=relu(s)^2 | 3 y=sigmoid(s) | 4 y=(y+gate*s)*scale
// ---------------------------------------------------------------------------
template <int NCOLS, int MODE>
__global__ void k_gemv_wmma(const float* __restrict__ W,
                            const float* __restrict__ x,
                            float* __restrict__ y,
                            const float* __restrict__ gate,
                            int nrows, float scale) {
    __shared__ __align__(32) _Float16 xs[NCOLS];   // fp16 copy of x (<= 8 KB)
    __shared__ float red[8][16];                   // per-wave partial row sums

    const int tid = threadIdx.x;
    // Vectorized fp16 staging of x: float4 loads + packed pair converts.
#pragma unroll
    for (int j = tid * 8; j < NCOLS; j += 256 * 8) {
        const float4 a0 = *(const float4*)(x + j);
        const float4 a1 = *(const float4*)(x + j + 4);
        h2* dst = (h2*)(xs + j);
        dst[0] = __builtin_amdgcn_cvt_pkrtz(a0.x, a0.y);
        dst[1] = __builtin_amdgcn_cvt_pkrtz(a0.z, a0.w);
        dst[2] = __builtin_amdgcn_cvt_pkrtz(a1.x, a1.y);
        dst[3] = __builtin_amdgcn_cvt_pkrtz(a1.z, a1.w);
    }
    __syncthreads();

    const int wave = tid >> 5;
    const int lane = tid & 31;
    const int rowInTile = lane & 15;
    const int hi = lane >> 4;                 // K half-select per ISA A layout
    const int row_base = blockIdx.x * 16;
    const int row = row_base + rowInTile;
    const bool valid = row < nrows;
    const float* rowp = W + (size_t)row * (size_t)NCOLS;

    constexpr int KPER = NCOLS >> 3;          // K per wave (8-way split-K)
    const int kbeg = wave * KPER;

    v8f c = {};
#pragma unroll 4
    for (int kk0 = 0; kk0 < KPER; kk0 += 32) {
        const int k0 = kbeg + kk0;
        // --- A operand: 16 fp32 weights/lane -> 8 packed-pair converts,
        //     contiguous K order matching B (lane<16: K 0..15; hi: K 16..31).
        union { v16h v; h2 p[8]; } ua;
        if (valid) {
            const float4* p0 = (const float4*)(rowp + k0 + hi * 16);
            const float4* p1 = (const float4*)(rowp + k0 + hi * 16 + 8);
            const float4 q0 = p0[0], q1 = p0[1];
            const float4 q2 = p1[0], q3 = p1[1];
            ua.p[0] = __builtin_amdgcn_cvt_pkrtz(q0.x, q0.y);
            ua.p[1] = __builtin_amdgcn_cvt_pkrtz(q0.z, q0.w);
            ua.p[2] = __builtin_amdgcn_cvt_pkrtz(q1.x, q1.y);
            ua.p[3] = __builtin_amdgcn_cvt_pkrtz(q1.z, q1.w);
            ua.p[4] = __builtin_amdgcn_cvt_pkrtz(q2.x, q2.y);
            ua.p[5] = __builtin_amdgcn_cvt_pkrtz(q2.z, q2.w);
            ua.p[6] = __builtin_amdgcn_cvt_pkrtz(q3.x, q3.y);
            ua.p[7] = __builtin_amdgcn_cvt_pkrtz(q3.z, q3.w);
            if (kk0 + 128 < KPER) __builtin_prefetch(rowp + k0 + 128, 0, 0);
        } else {
#pragma unroll
            for (int j = 0; j < 16; ++j) ua.v[j] = (_Float16)0.f;
        }
        // --- B operand: x[k0 + hi*16 .. +15] replicated across all 16 columns
        //     (32-byte aligned -> wide ds loads, static offsets).
        const v16h bv = *(const v16h*)(xs + k0 + hi * 16);

        c = __builtin_amdgcn_wmma_f32_16x16x32_f16(
                /*neg_a=*/false, ua.v, /*neg_b=*/false, bv,
                /*c_mod=*/(short)0, c, /*reuse_a=*/false, /*reuse_b=*/false);
    }

    // All 16 D columns are identical; lane 0 holds rows 0..7, lane 16 rows 8..15.
    if (lane == 0) {
#pragma unroll
        for (int rr = 0; rr < 8; ++rr) red[wave][rr] = c[rr];
    } else if (lane == 16) {
#pragma unroll
        for (int rr = 0; rr < 8; ++rr) red[wave][8 + rr] = c[rr];
    }
    __syncthreads();

    if (tid < 16) {
        float s = 0.f;
#pragma unroll
        for (int w8 = 0; w8 < 8; ++w8) s += red[w8][tid];
        const int orow = row_base + tid;
        if (orow < nrows) {
            float out;
            if      (MODE == 0) out = s;
            else if (MODE == 1) out = y[orow] + s;
            else if (MODE == 2) { const float r = fmaxf(s, 0.f); out = r * r; }
            else if (MODE == 3) out = 1.f / (1.f + expf(-s));
            else                out = (y[orow] + gate[orow] * s) * scale;
            y[orow] = out;
        }
    }
}

// ---------------------------------------------------------------------------
// Host-side orchestration (stream-ordered; graph-capture safe).
// ---------------------------------------------------------------------------
extern "C" void kernel_launch(void* const* d_in, const int* in_sizes, int n_in,
                              void* d_out_v, int out_size, void* d_ws, size_t ws_size,
                              hipStream_t stream) {
    (void)in_sizes; (void)n_in; (void)out_size; (void)ws_size;

    const float* token_embd = (const float*)d_in[0];
    const float* state      = (const float*)d_in[1];
    const float* emb_ln_w   = (const float*)d_in[2];
    const float* emb_ln_b   = (const float*)d_in[3];
    const float* ln1_w      = (const float*)d_in[4];
    const float* ln1_b      = (const float*)d_in[5];
    const float* ln2_w      = (const float*)d_in[6];
    const float* ln2_b      = (const float*)d_in[7];
    const float* att_tmk    = (const float*)d_in[8];
    const float* att_tmv    = (const float*)d_in[9];
    const float* att_tmr    = (const float*)d_in[10];
    const float* att_tf     = (const float*)d_in[11];
    const float* att_td     = (const float*)d_in[12];
    const float* att_kw     = (const float*)d_in[13];
    const float* att_vw     = (const float*)d_in[14];
    const float* att_rw     = (const float*)d_in[15];
    const float* att_ow     = (const float*)d_in[16];
    const float* ffn_tmk    = (const float*)d_in[17];
    const float* ffn_tmr    = (const float*)d_in[18];
    const float* ffn_kw     = (const float*)d_in[19];
    const float* ffn_vw     = (const float*)d_in[20];
    const float* ffn_rw     = (const float*)d_in[21];
    const float* out_ln_w   = (const float*)d_in[22];
    const float* out_ln_b   = (const float*)d_in[23];
    const float* head_w     = (const float*)d_in[24];

    float* logits    = (float*)d_out_v;            // [V]
    float* out_state = logits + VOCAB;             // [L,5,D]

    float* ws  = (float*)d_ws;
    float* x   = ws;                 // [D] residual stream
    float* xk  = ws + 1  * D_EMB;
    float* xv  = ws + 2  * D_EMB;
    float* xr  = ws + 3  * D_EMB;
    float* kk  = ws + 4  * D_EMB;
    float* vv  = ws + 5  * D_EMB;
    float* rr  = ws + 6  * D_EMB;
    float* twk = ws + 7  * D_EMB;    // wkv output
    float* xk2 = ws + 8  * D_EMB;
    float* xr2 = ws + 9  * D_EMB;
    float* r2  = ws + 10 * D_EMB;
    float* k2  = ws + 11 * D_EMB;    // [F]
    float* xh  = ws + 15 * D_EMB;

    const dim3 blk(256);
    const int gD = D_EMB / 16;
    const int gF = F_HID / 16;
    const int gV = (VOCAB + 15) / 16;

    // encoder: x = LN(token_embd)
    k_ln_mix<<<1, blk, 0, stream>>>(token_embd, D_EMB, emb_ln_w, emb_ln_b,
                                    nullptr, nullptr, nullptr, nullptr,
                                    x, nullptr, nullptr, nullptr);

    for (int i = 0; i < N_LAYER; ++i) {
        const float* st  = state     + (size_t)i * 5 * D_EMB;
        float*       ost = out_state + (size_t)i * 5 * D_EMB;
        const size_t mo  = (size_t)i * D_EMB * D_EMB;
        const size_t fo  = (size_t)i * (size_t)F_HID * D_EMB;

        // ---- time mixing ----
        k_ln_mix<<<1, blk, 0, stream>>>(x, D_EMB, ln1_w + i * D_EMB, ln1_b + i * D_EMB,
                                        st + D_EMB,                       // s_att = st[1]
                                        att_tmk + i * D_EMB, att_tmv + i * D_EMB,
                                        att_tmr + i * D_EMB,
                                        ost + D_EMB,                      // new_st[1] = xn
                                        xk, xv, xr);
        k_gemv_wmma<D_EMB, 0><<<gD, blk, 0, stream>>>(att_kw + mo, xk, kk, nullptr, D_EMB, 1.f);
        k_gemv_wmma<D_EMB, 0><<<gD, blk, 0, stream>>>(att_vw + mo, xv, vv, nullptr, D_EMB, 1.f);
        k_gemv_wmma<D_EMB, 3><<<gD, blk, 0, stream>>>(att_rw + mo, xr, rr, nullptr, D_EMB, 1.f); // sigmoid
        k_wkv<<<1, blk, 0, stream>>>(kk, vv, rr,
                                     att_tf + i * D_EMB, att_td + i * D_EMB,
                                     st + 2 * D_EMB, st + 3 * D_EMB, st + 4 * D_EMB,
                                     twk, ost + 2 * D_EMB, ost + 3 * D_EMB, ost + 4 * D_EMB);
        k_gemv_wmma<D_EMB, 1><<<gD, blk, 0, stream>>>(att_ow + mo, twk, x, nullptr, D_EMB, 1.f); // x += ow@t

        // ---- channel mixing ----
        k_ln_mix<<<1, blk, 0, stream>>>(x, D_EMB, ln2_w + i * D_EMB, ln2_b + i * D_EMB,
                                        st,                                // s_ffn = st[0]
                                        ffn_tmk + i * D_EMB, nullptr,
                                        ffn_tmr + i * D_EMB,
                                        ost,                               // new_st[0] = xn2
                                        xk2, nullptr, xr2);
        k_gemv_wmma<D_EMB, 2><<<gF, blk, 0, stream>>>(ffn_kw + fo, xk2, k2, nullptr, F_HID, 1.f); // relu^2
        k_gemv_wmma<D_EMB, 3><<<gD, blk, 0, stream>>>(ffn_rw + mo, xr2, r2, nullptr, D_EMB, 1.f); // sigmoid
        const float scale = (((i + 1) % 6) == 0) ? 0.5f : 1.f;             // RESCALE
        k_gemv_wmma<F_HID, 4><<<gD, blk, 0, stream>>>(ffn_vw + fo, k2, x, r2, D_EMB, scale);
    }

    // decoder: logits = head_w @ LN(x)
    k_ln_mix<<<1, blk, 0, stream>>>(x, D_EMB, out_ln_w, out_ln_b,
                                    nullptr, nullptr, nullptr, nullptr,
                                    xh, nullptr, nullptr, nullptr);
    k_gemv_wmma<D_EMB, 0><<<gV, blk, 0, stream>>>(head_w, xh, logits, nullptr, VOCAB, 1.f);
}
